// CrossLayerTranscoder_10350871183814
// MI455X (gfx1250) — matrix operand order repaired and checked by reading, not compiled, not run on hardware
//
#include <hip/hip_runtime.h>

// ---------------------------------------------------------------------------
// Cross-layer transcoder on MI455X (gfx1250, wave32, WMMA).
//   encode:  pre = relu(X @ W_enc + b_enc)        -> bf16 workspace (in-place topk)
//   topk:    keep top-64 per row (radix-select on bf16 bits), zero the rest
//   decode:  out[j] = b_dec[j] + sum_{i<=j} acts[i] @ W_dec[i,j]   (dense bf16 WMMA)
// ---------------------------------------------------------------------------

#define N_LAYERS 12
#define D_IN     768
#define D_DICT   4096
#define D_OUT    768
#define BATCH    2048
#define TOP_K    64

typedef unsigned short u16;
typedef unsigned int   u32;

typedef __bf16 bf16;
typedef bf16  v16bf __attribute__((ext_vector_type(16)));
typedef bf16  v8bf  __attribute__((ext_vector_type(8)));
typedef float v8f   __attribute__((ext_vector_type(8)));

// LDS tile row stride (in bf16 elements): 32 K-values + 8 pad, keeps 16B
// alignment for the b128 fragment reads while breaking power-of-2 banking.
#define LDS_STRIDE 40

__device__ __forceinline__ u16 f2bf(float f) {
    u32 u = __builtin_bit_cast(u32, f);
    u += 0x7FFFu + ((u >> 16) & 1u);   // round-to-nearest-even
    return (u16)(u >> 16);
}

__device__ __forceinline__ u32 pack2bf(float lo, float hi) {
    return (u32)f2bf(lo) | ((u32)f2bf(hi) << 16);
}

__device__ __forceinline__ v16bf cat16(v8bf lo, v8bf hi) {
    return __builtin_shufflevector(lo, hi, 0, 1, 2, 3, 4, 5, 6, 7,
                                   8, 9, 10, 11, 12, 13, 14, 15);
}

// A-matrix 16x32 bf16 fragment (ISA 7.12.2): lane m = lane&15,
// koff = lane<16 ? 0 : 8; element e -> K = koff + e (e<8) / koff + 8 + e (e>=8).
__device__ __forceinline__ v16bf load_afrag(const bf16* s, int row0, int lane) {
    const int koff = (lane & 16) ? 8 : 0;
    const bf16* p  = s + (row0 + (lane & 15)) * LDS_STRIDE;
    v8bf lo = *(const v8bf*)(p + koff);
    v8bf hi = *(const v8bf*)(p + koff + 16);
    return cat16(lo, hi);
}

// B-matrix 32x16 bf16 fragment: lane n = lane&15, koff = lane<16 ? 0 : 8;
// element e -> K = koff + e (e<8) / koff + 8 + e (e>=8). LDS tile is stored N-major.
__device__ __forceinline__ v16bf load_bfrag(const bf16* s, int row0, int lane) {
    const int kb  = (lane & 16) ? 8 : 0;
    const bf16* p = s + (row0 + (lane & 15)) * LDS_STRIDE;
    v8bf lo = *(const v8bf*)(p + kb);
    v8bf hi = *(const v8bf*)(p + kb + 16);
    return cat16(lo, hi);
}

__device__ __forceinline__ v8f wmma_bf16(v16bf a, v16bf b, v8f c) {
    return __builtin_amdgcn_wmma_f32_16x16x32_bf16(
        /*neg_a=*/false, a, /*neg_b=*/false, b,
        /*c_mod=*/(short)0, c, /*reuse_a=*/false, /*reuse_b=*/false);
}

// Hoisted fragment loads + back-to-back WMMA chain: one s_wait_dscnt covers
// all eight v_wmma issues instead of a wait per fm.
__device__ __forceinline__ void mma_tile(const bf16* sA, const bf16* sB,
                                         int wm, int wn, int lane,
                                         v8f acc[4][2]) {
    v16bf afr[4];
    v16bf bfr[2];
    #pragma unroll
    for (int fm = 0; fm < 4; ++fm) afr[fm] = load_afrag(sA, wm + fm * 16, lane);
    #pragma unroll
    for (int fn = 0; fn < 2; ++fn) bfr[fn] = load_bfrag(sB, wn + fn * 16, lane);
    #pragma unroll
    for (int fm = 0; fm < 4; ++fm)
        #pragma unroll
        for (int fn = 0; fn < 2; ++fn)
            acc[fm][fn] = wmma_bf16(afr[fm], bfr[fn], acc[fm][fn]);
}

// ---------------------------------------------------------------------------
// Kernel 1: encode.  grid = (BATCH/128, D_DICT/128, N_LAYERS), block = 256.
// Each block: 128x128 output tile; 8 waves in a 2x4 grid, 64x32 per wave.
// ---------------------------------------------------------------------------
__global__ __launch_bounds__(256)
void encode_wmma_bf16(const float* __restrict__ X,     // [12,2048,768]
                      const float* __restrict__ W,     // [12,768,4096]
                      const float* __restrict__ bias,  // [12,4096]
                      u16* __restrict__ pre)           // [12,2048,4096] bf16
{
    const int layer = blockIdx.z;
    X    += (size_t)layer * BATCH * D_IN;
    W    += (size_t)layer * D_IN * D_DICT;
    bias += (size_t)layer * D_DICT;
    pre  += (size_t)layer * BATCH * D_DICT;

    const int m0 = blockIdx.x * 128;
    const int n0 = blockIdx.y * 128;

    __shared__ bf16 sA[128 * LDS_STRIDE];
    __shared__ bf16 sB[128 * LDS_STRIDE];

    const int t    = threadIdx.x;
    const int wave = t >> 5;
    const int lane = t & 31;
    const int wm   = (wave >> 2) * 64;   // 0 or 64
    const int wn   = (wave & 3) * 32;    // 0,32,64,96

    v8f acc[4][2] = {};

    for (int k0 = 0; k0 < D_IN; k0 += 32) {
        __syncthreads();
        // Stage X tile [128 rows][32 K] fp32 -> bf16, M-major, packed b64 stores.
        #pragma unroll
        for (int it = 0; it < 4; ++it) {
            int e = t + it * 256;            // 0..1023, 4 floats each
            int r = e >> 3;
            int c = (e & 7) * 4;
            float4 f = *(const float4*)&X[(size_t)(m0 + r) * D_IN + k0 + c];
            uint2 d;
            d.x = pack2bf(f.x, f.y);
            d.y = pack2bf(f.z, f.w);
            *(uint2*)((u32*)sA + r * (LDS_STRIDE / 2) + (c >> 1)) = d;
        }
        // Stage W tile [32 K][128 N] -> bf16, N-major.  Each thread takes a
        // (k, k+1) pair of one column n: two coalesced b32 loads across lanes,
        // one packed b32 LDS store (no d16_hi scatter).
        #pragma unroll
        for (int it = 0; it < 8; ++it) {
            int e  = t + it * 256;           // 0..2047
            int kp = e >> 7;                 // 0..15 -> k = 2*kp
            int n  = e & 127;
            const float* wp = &W[(size_t)(k0 + 2 * kp) * D_DICT + n0 + n];
            float f0 = wp[0];
            float f1 = wp[D_DICT];
            ((u32*)sB)[n * (LDS_STRIDE / 2) + kp] = pack2bf(f0, f1);
        }
        __syncthreads();

        mma_tile(sA, sB, wm, wn, lane, acc);
    }

    // Epilogue: C/D layout (ISA 7.12.2): lane -> N=lane&15, Mbase=(lane>>4)*8.
    const int nl = lane & 15;
    const int mb = (lane >> 4) * 8;
    #pragma unroll
    for (int fm = 0; fm < 4; ++fm) {
        #pragma unroll
        for (int fn = 0; fn < 2; ++fn) {
            const int n = n0 + wn + fn * 16 + nl;
            const float bv = bias[n];
            #pragma unroll
            for (int v = 0; v < 8; ++v) {
                const int m = m0 + wm + fm * 16 + mb + v;
                float x = acc[fm][fn][v] + bv;
                x = fmaxf(x, 0.0f);
                pre[(size_t)m * D_DICT + n] = f2bf(x);
            }
        }
    }
}

// ---------------------------------------------------------------------------
// Kernel 2: in-place top-64 per row.  grid = N_LAYERS*BATCH, block = 256.
// Post-ReLU values are >= 0, so bf16 bit patterns order like the values:
// radix-select the 64th-largest bit pattern, then mask with tie budgeting.
// ---------------------------------------------------------------------------
__global__ __launch_bounds__(256)
void topk64_inplace(u16* __restrict__ pre)   // [12*2048, 4096] bf16
{
    const size_t base = (size_t)blockIdx.x * D_DICT;
    const int t = threadIdx.x;

    u16 v[16];
    #pragma unroll
    for (int e = 0; e < 16; ++e) v[e] = pre[base + t * 16 + e];

    __shared__ u32 s_cnt;
    __shared__ u32 s_tie;

    u32 prefix = 0;
    for (int bit = 14; bit >= 0; --bit) {       // bit 15 = sign, always 0
        const u32 cand = prefix | (1u << bit);
        if (t == 0) s_cnt = 0;
        __syncthreads();
        u32 c = 0;
        #pragma unroll
        for (int e = 0; e < 16; ++e) c += ((u32)v[e] >= cand) ? 1u : 0u;
        if (c) atomicAdd(&s_cnt, c);
        __syncthreads();
        const u32 total = s_cnt;
        __syncthreads();                         // protect s_cnt reset next iter
        if (total >= TOP_K) prefix = cand;
    }

    // Count strictly-greater, then budget ties at the threshold.
    if (t == 0) { s_cnt = 0; s_tie = 0; }
    __syncthreads();
    u32 cg = 0;
    #pragma unroll
    for (int e = 0; e < 16; ++e) cg += ((u32)v[e] > prefix) ? 1u : 0u;
    if (cg) atomicAdd(&s_cnt, cg);
    __syncthreads();
    const u32 n_gt  = s_cnt;
    const u32 slots = (n_gt < TOP_K) ? (TOP_K - n_gt) : 0u;

    #pragma unroll
    for (int e = 0; e < 16; ++e) {
        const u16 x = v[e];
        bool keep = ((u32)x > prefix);
        if (!keep && (u32)x == prefix) {
            const u32 id = atomicAdd(&s_tie, 1u);
            keep = (id < slots);
        }
        pre[base + t * 16 + e] = keep ? x : (u16)0;
    }
}

// ---------------------------------------------------------------------------
// Kernel 3: triangular decode.  grid = (BATCH/128, D_OUT/128, N_LAYERS).
// out[j] = b_dec[j] + sum_{i<=j} acts_bf16[i] @ bf16(W_dec[i,j])
// The acts tile is a raw bf16 copy -> use the CDNA5 async global->LDS path
// (GLOBAL_LOAD_ASYNC_TO_LDS_B128, tracked on ASYNCcnt), bypassing VGPRs.
// ---------------------------------------------------------------------------
__global__ __launch_bounds__(256)
void decode_wmma_bf16(const u16* __restrict__ acts,   // [12,2048,4096] bf16
                      const float* __restrict__ Wd,   // [12,12,4096,768]
                      const float* __restrict__ bd,   // [12,768]
                      float* __restrict__ out)        // [12,2048,768]
{
    const int j  = blockIdx.z;
    const int m0 = blockIdx.x * 128;
    const int n0 = blockIdx.y * 128;

    __shared__ bf16 sA[128 * LDS_STRIDE];
    __shared__ bf16 sB[128 * LDS_STRIDE];

    const int t    = threadIdx.x;
    const int wave = t >> 5;
    const int lane = t & 31;
    const int wm   = (wave >> 2) * 64;
    const int wn   = (wave & 3) * 32;

    v8f acc[4][2] = {};

    for (int i = 0; i <= j; ++i) {
        const u16*   A = acts + (size_t)i * BATCH * D_DICT;
        const float* W = Wd + ((size_t)i * N_LAYERS + j) * (size_t)D_DICT * D_OUT;

        for (int k0 = 0; k0 < D_DICT; k0 += 32) {
            __syncthreads();
            // Stage acts tile: async 16B global->LDS copies (GV mode), M-major.
            #pragma unroll
            for (int it = 0; it < 2; ++it) {
                int e = t + it * 256;          // 0..511, 8 bf16 each
                int r = e >> 2;
                int c = (e & 3) * 8;
                const u16* src = &A[(size_t)(m0 + r) * D_DICT + k0 + c];
                u32 dst = (u32)(size_t)((const u16*)sA + r * LDS_STRIDE + c);
                asm volatile("global_load_async_to_lds_b128 %0, %1, off"
                             :: "v"(dst), "v"(src) : "memory");
            }
            // Stage W_dec tile [32 K][128 N] -> bf16, N-major, packed stores.
            #pragma unroll
            for (int it = 0; it < 8; ++it) {
                int e  = t + it * 256;
                int kp = e >> 7;
                int n  = e & 127;
                const float* wp = &W[(size_t)(k0 + 2 * kp) * D_OUT + n0 + n];
                float f0 = wp[0];
                float f1 = wp[D_OUT];
                ((u32*)sB)[n * (LDS_STRIDE / 2) + kp] = pack2bf(f0, f1);
            }
            // Prefetch next K-slab of W_dec into GL2 (global_prefetch_b8).
            if (k0 + 32 < D_DICT)
                __builtin_prefetch(&W[(size_t)(k0 + 32) * D_OUT + n0 + ((t & 63) * 4)], 0, 1);
            // Async copies complete on ASYNCcnt (not covered by barrier waits).
            asm volatile("s_wait_asynccnt 0x0" ::: "memory");
            __syncthreads();

            mma_tile(sA, sB, wm, wn, lane, acc);
        }
    }

    const int nl = lane & 15;
    const int mb = (lane >> 4) * 8;
    #pragma unroll
    for (int fm = 0; fm < 4; ++fm) {
        #pragma unroll
        for (int fn = 0; fn < 2; ++fn) {
            const int n = n0 + wn + fn * 16 + nl;
            const float bv = bd[(size_t)j * D_OUT + n];
            #pragma unroll
            for (int v = 0; v < 8; ++v) {
                const int m = m0 + wm + fm * 16 + mb + v;
                out[((size_t)j * BATCH + m) * D_OUT + n] = acc[fm][fn][v] + bv;
            }
        }
    }
}

// ---------------------------------------------------------------------------
extern "C" void kernel_launch(void* const* d_in, const int* in_sizes, int n_in,
                              void* d_out, int out_size, void* d_ws, size_t ws_size,
                              hipStream_t stream) {
    const float* inputs = (const float*)d_in[0];   // [12,2048,768]
    const float* W_enc  = (const float*)d_in[1];   // [12,768,4096]
    const float* b_enc  = (const float*)d_in[2];   // [12,4096]
    const float* W_dec  = (const float*)d_in[3];   // [12,12,4096,768]
    const float* b_dec  = (const float*)d_in[4];   // [12,768]
    float* out = (float*)d_out;                    // [12,2048,768]

    // Workspace: bf16 pre-activations / sparse acts (in-place top-k): 201 MB.
    u16* pre = (u16*)d_ws;

    encode_wmma_bf16<<<dim3(BATCH / 128, D_DICT / 128, N_LAYERS), 256, 0, stream>>>(
        inputs, W_enc, b_enc, pre);

    topk64_inplace<<<dim3(N_LAYERS * BATCH), 256, 0, stream>>>(pre);

    decode_wmma_bf16<<<dim3(BATCH / 128, D_OUT / 128, N_LAYERS), 256, 0, stream>>>(
        pre, W_dec, b_dec, out);
}
